// ModulatedConv2dWithUpsample_18262200943050
// MI455X (gfx1250) — matrix operand order repaired and checked by reading, not compiled, not run on hardware
//
#include <hip/hip_runtime.h>
#include <hip/hip_bf16.h>
#include <stddef.h>
#include <stdint.h>

// ModulatedConv2dWithUpsample for MI455X (gfx1250).
// convT(stride2,3x3)+blur(4x4) fused into 4 phase-convs of 3x3 (upfirdn
// identity) -> implicit GEMM on v_wmma_f32_16x16x32_bf16, f32 accumulate.
//
// Pipeline:
//   1) mod_kernel  : s = style @ (mod_w/sqrt(512))^T + bias           [8,512]
//   2) prep_kernel : demod + fold blur -> Keff[b][4][9][co][ci] bf16  (75.5MB)
//   3) tr_kernel   : Xt[b][1+y][1+x][ci] = bf16(input[b][ci][y][x]),
//                    zero halo -> unconditional B-fragment loads      (35.7MB)
//   4) gemm_kernel : no LDS, no branches in hot loop; A and B WMMA
//                    fragments are K-contiguous 16B global chunks (L2-hot).
//
// Workspace: 16KB (s) + 75.5MB (Keff) + 35.7MB (Xt padded) ~= 112MB.

#define B_    8
#define CIN   512
#define COUT  256
#define H_    64
#define W_    64
#define WDIM  512
#define HP    66          // padded spatial dim (1-px zero halo)

typedef __attribute__((ext_vector_type(16))) __bf16       v16bf;
typedef __attribute__((ext_vector_type(8)))  float        v8f;
typedef __attribute__((ext_vector_type(4)))  unsigned int u32x4;

union Frag {              // 32 bytes: one WMMA A/B operand (8 VGPRs)
    u32x4 q[2];
    v16bf v;
};

__device__ __forceinline__ unsigned short f2bf(float f) {
    unsigned int u = __float_as_uint(f);
    unsigned int r = u + 0x7FFFu + ((u >> 16) & 1u);   // round-to-nearest-even
    return (unsigned short)(r >> 16);
}

// ---------------------------------------------------------------------------
// Kernel 1: s[b][ci] = (style[b] . mod_weight[ci]) / sqrt(512) + mod_bias[ci]
// ---------------------------------------------------------------------------
__global__ __launch_bounds__(256) void mod_kernel(
    const float* __restrict__ style, const float* __restrict__ mw,
    const float* __restrict__ mb, float* __restrict__ s)
{
    int idx = blockIdx.x * 256 + threadIdx.x;     // 0..4095
    int ci = idx & (CIN - 1);
    int b  = idx >> 9;
    const float ms = 0.04419417382415922f;        // 1/sqrt(512)
    float acc = 0.0f;
    const float* st = style + (size_t)b * WDIM;
    const float* wr = mw + (size_t)ci * WDIM;
#pragma unroll 8
    for (int j = 0; j < WDIM; ++j) acc += st[j] * wr[j];
    s[idx] = acc * ms + mb[ci];
}

// ---------------------------------------------------------------------------
// Kernel 2: modulate + demodulate, fold blur into 4 phase 3x3 kernels.
// Keff layout: [b][phase(4)][tap(9)][co][ci] bf16
// ---------------------------------------------------------------------------
__constant__ float Ctab[2][3][3] = {
    { {0.00f, 0.25f, 0.75f},   // phase 0, d=-1
      {0.75f, 0.75f, 0.25f},   //          d= 0
      {0.25f, 0.00f, 0.00f} }, //          d=+1
    { {0.00f, 0.00f, 0.25f},   // phase 1, d=-1
      {0.25f, 0.75f, 0.75f},   //          d= 0
      {0.75f, 0.25f, 0.00f} }  //          d=+1
};

__global__ __launch_bounds__(256) void prep_kernel(
    const float* __restrict__ weight,   // [Cout][Cin][3][3]
    const float* __restrict__ s,        // [B][Cin]
    unsigned short* __restrict__ keff)
{
    __shared__ float red[256];
    int b  = blockIdx.x >> 8;
    int co = blockIdx.x & 255;
    int tid = threadIdx.x;
    const float scale = 0.014731391274719739f;   // 1/sqrt(512*9)

    float wloc[2][9];
    float local = 0.0f;
#pragma unroll
    for (int p = 0; p < 2; ++p) {
        int ci = tid + p * 256;
        float sv = s[(size_t)b * CIN + ci];
        const float* wr = weight + ((size_t)co * CIN + ci) * 9;
#pragma unroll
        for (int j = 0; j < 9; ++j) {
            float wv = scale * wr[j] * sv;
            wloc[p][j] = wv;
            local += wv * wv;
        }
    }
    red[tid] = local;
    __syncthreads();
#pragma unroll
    for (int off = 128; off > 0; off >>= 1) {
        if (tid < off) red[tid] += red[tid + off];
        __syncthreads();
    }
    float demod = rsqrtf(red[0] + 1e-8f);

#pragma unroll
    for (int p = 0; p < 2; ++p) {
        int ci = tid + p * 256;
        float wm[9];
#pragma unroll
        for (int j = 0; j < 9; ++j) wm[j] = wloc[p][j] * demod;
#pragma unroll
        for (int py = 0; py < 2; ++py)
#pragma unroll
        for (int px = 0; px < 2; ++px)
#pragma unroll
        for (int dy = 0; dy < 3; ++dy)
#pragma unroll
        for (int dx = 0; dx < 3; ++dx) {
            float acc = 0.0f;
#pragma unroll
            for (int jy = 0; jy < 3; ++jy)
#pragma unroll
            for (int jx = 0; jx < 3; ++jx)
                acc += Ctab[py][dy][jy] * Ctab[px][dx][jx] * wm[jy * 3 + jx];
            int ph  = py * 2 + px;
            int tap = dy * 3 + dx;
            size_t o = ((((size_t)b * 4 + ph) * 9 + tap) * COUT + co) * CIN + ci;
            keff[o] = f2bf(acc);
        }
    }
}

// ---------------------------------------------------------------------------
// Kernel 3: transpose+convert with zero halo:
//   Xt[b][1+y][1+x][ci] = bf16(in[b][ci][y][x]);  halo rows/cols = 0
// One block per (b,y): LDS 64x64 tiles, coalesced in, 32B-chunk out.
// ---------------------------------------------------------------------------
__global__ __launch_bounds__(256) void tr_kernel(
    const float* __restrict__ in, unsigned short* __restrict__ xt)
{
    __shared__ unsigned short tile[64][72];      // pad to spread banks
    int b = blockIdx.x >> 6;
    int y = blockIdx.x & 63;
    int t = threadIdx.x;
    int x  = t & 63, k0 = t >> 6;                // load mapping
    int xw = t >> 2, kq = (t & 3) * 16;          // store mapping

    unsigned short* rowOut = xt + ((size_t)b * HP + (y + 1)) * HP * CIN;

    for (int cb = 0; cb < CIN; cb += 64) {
        __syncthreads();
#pragma unroll
        for (int k = k0; k < 64; k += 4)
            tile[k][x] = f2bf(in[(((size_t)b * CIN + cb + k) * H_ + y) * W_ + x]);
        __syncthreads();
        union { u32x4 q[2]; unsigned short h[16]; } r;
#pragma unroll
        for (int i = 0; i < 16; ++i) r.h[i] = tile[kq + i][xw];
        u32x4* dst = (u32x4*)(rowOut + (size_t)(xw + 1) * CIN + cb + kq);
        dst[0] = r.q[0];
        dst[1] = r.q[1];
    }

    const u32x4 z = {0u, 0u, 0u, 0u};
    // left/right halo columns of this row: 2 * 512 bf16 = 128 x 16B chunks
    if (t < 128) {
        int col = (t >> 6) ? (HP - 1) : 0;
        ((u32x4*)(rowOut + (size_t)col * CIN))[t & 63] = z;
    }
    // top/bottom halo rows: 66*512 bf16 = 4224 x 16B chunks
    if (y == 0 || y == 63) {
        unsigned short* hrow =
            xt + ((size_t)b * HP + (y == 0 ? 0 : (HP - 1))) * HP * CIN;
        for (int i = t; i < (HP * CIN) / 8; i += 256)
            ((u32x4*)hrow)[i] = z;
    }
}

// ---------------------------------------------------------------------------
// Kernel 4: implicit GEMM, fragments direct from global; branch-free hot loop.
// Block: 256 thr = 8 wave32, all along M (Cout=256). Wave tile: 32(M) x 64(N).
// Grid = B * 4 phases * 64 rows = 2048 blocks.
// ---------------------------------------------------------------------------
__global__ __launch_bounds__(256) void gemm_kernel(
    const unsigned short* __restrict__ xt,    // [B][66][66][512] bf16, halo=0
    const unsigned short* __restrict__ keff,  // [B][4][9][256][512] bf16
    float* __restrict__ out)                  // [B][Cout][128][128] f32
{
    int bid = blockIdx.x;
    int u   = bid & 63;
    int ph  = (bid >> 6) & 3;
    int b   = bid >> 8;
    int py  = ph >> 1, px = ph & 1;

    int tid  = threadIdx.x;
    int lane = tid & 31;
    int w    = tid >> 5;                 // wave id 0..7 -> M rows w*32..w*32+31
    int l16  = lane & 15;
    bool hiL = lane >= 16;
    // ISA 7.12.2 wave32 layouts:
    //  A 16x32: lanes<16 hold K 0..7 & 16..23, lanes>=16 hold K 8..15 & 24..31
    //  B 32x16: lanes<16 hold K 0..7 & 16..23, lanes>=16 hold K 8..15 & 24..31; N = lane%16
    int akOff = hiL ? 8 : 0;
    int bkOff = hiL ? 8 : 0;

    v8f acc[2][4];
#pragma unroll
    for (int i = 0; i < 2; ++i)
#pragma unroll
        for (int j = 0; j < 4; ++j)
#pragma unroll
            for (int r = 0; r < 8; ++r) acc[i][j][r] = 0.0f;

    const unsigned short* aPh =
        keff + (((size_t)b * 4 + ph) * 9) * COUT * CIN;     // + tap*COUT*CIN

    for (int dy = 0; dy < 3; ++dy) {
        int iu = u + dy;                                    // padded row index
        const unsigned short* bRowBase =
            xt + ((size_t)b * HP + iu) * HP * CIN + bkOff;
        for (int dx = 0; dx < 3; ++dx) {
            int tap = dy * 3 + dx;
            const unsigned short* aR[2];
#pragma unroll
            for (int mi = 0; mi < 2; ++mi) {
                int co = w * 32 + mi * 16 + l16;
                aR[mi] = aPh + ((size_t)tap * COUT + co) * CIN + akOff;
            }
            const unsigned short* bR[4];
#pragma unroll
            for (int ni = 0; ni < 4; ++ni) {
                int ix = ni * 16 + l16 + dx;                // padded col index
                bR[ni] = bRowBase + (size_t)ix * CIN;
            }

#pragma unroll
            for (int kb = 0; kb < CIN; kb += 32) {
                Frag fa[2], fb[4];
#pragma unroll
                for (int mi = 0; mi < 2; ++mi) {
                    fa[mi].q[0] = *(const u32x4*)(aR[mi] + kb);
                    fa[mi].q[1] = *(const u32x4*)(aR[mi] + kb + 16);
                }
#pragma unroll
                for (int ni = 0; ni < 4; ++ni) {
                    fb[ni].q[0] = *(const u32x4*)(bR[ni] + kb);
                    fb[ni].q[1] = *(const u32x4*)(bR[ni] + kb + 16);
                }
#pragma unroll
                for (int mi = 0; mi < 2; ++mi)
#pragma unroll
                    for (int ni = 0; ni < 4; ++ni)
                        acc[mi][ni] = __builtin_amdgcn_wmma_f32_16x16x32_bf16(
                            false, fa[mi].v, false, fb[ni].v,
                            (short)0, acc[mi][ni], false, false);
            }
        }
    }

    // store: C layout VGPR r -> M = r (+8 for lanes>=16), N = lane%16
    int y = 2 * u + py;
#pragma unroll
    for (int mi = 0; mi < 2; ++mi)
#pragma unroll
    for (int ni = 0; ni < 4; ++ni) {
        int n  = ni * 16 + l16;
        int x  = 2 * n + px;
        int mB = w * 32 + mi * 16 + (hiL ? 8 : 0);
#pragma unroll
        for (int r = 0; r < 8; ++r) {
            int co = mB + r;
            out[(((size_t)b * COUT + co) * 128 + y) * 128 + x] = acc[mi][ni][r];
        }
    }
}

// ---------------------------------------------------------------------------
extern "C" void kernel_launch(void* const* d_in, const int* in_sizes, int n_in,
                              void* d_out, int out_size, void* d_ws, size_t ws_size,
                              hipStream_t stream) {
    (void)in_sizes; (void)n_in; (void)out_size; (void)ws_size;
    const float* input      = (const float*)d_in[0];   // [8,512,64,64]
    const float* style      = (const float*)d_in[1];   // [8,512]
    const float* weight     = (const float*)d_in[2];   // [1,256,512,3,3]
    const float* mod_weight = (const float*)d_in[3];   // [512,512]
    const float* mod_bias   = (const float*)d_in[4];   // [512]
    float* out = (float*)d_out;                        // [8,256,128,128] f32

    float*          s    = (float*)d_ws;                                  // 16 KB
    unsigned short* keff = (unsigned short*)((char*)d_ws + 16384);        // 75.5 MB
    unsigned short* xt   = (unsigned short*)((char*)d_ws + 16384 + 75497472); // 35.7 MB

    mod_kernel <<<(B_ * CIN) / 256, 256, 0, stream>>>(style, mod_weight, mod_bias, s);
    prep_kernel<<<B_ * COUT,        256, 0, stream>>>(weight, s, keff);
    tr_kernel  <<<B_ * H_,          256, 0, stream>>>(input, xt);
    gemm_kernel<<<B_ * 4 * H_,      256, 0, stream>>>(xt, keff, out);
}
